// CoxLoss_69879117906541
// MI455X (gfx1250) — compile-verified, matrix operand order checked by reading, not run on hardware
//
#include <hip/hip_runtime.h>
#include <cmath>

typedef __attribute__((ext_vector_type(16))) _Float16 v16h;
typedef __attribute__((ext_vector_type(8)))  _Float16 v8h;
typedef __attribute__((ext_vector_type(8)))  float    v8f;
typedef __attribute__((ext_vector_type(4)))  float    v4f;

// Wave-per-tile Cox risk-sum via WMMA.
//  A[m][k] = (s[tile*16+m] <= s[jb+k]) ? sigmoid(theta[jb+k]) : 0   (f16)
//  B       = all ones (layout-invariant)
//  D[m][*] accumulates risk_sum[tile*16+m] in f32 across all K chunks.
__global__ void __launch_bounds__(256)
cox_risk_wmma_kernel(const float* __restrict__ survtime,
                     const float* __restrict__ censor,
                     const float* __restrict__ theta,
                     float* __restrict__ terms, int n)
{
    extern __shared__ char smem[];
    float*    s_s = reinterpret_cast<float*>(smem);                       // n f32
    _Float16* s_e = reinterpret_cast<_Float16*>(smem + (size_t)n * 4u);   // n f16

    const int tid = threadIdx.x;

    // Cooperative stage of the whole problem into LDS (192KB inputs -> 96KB LDS).
    for (int j = tid; j < n; j += blockDim.x) {
        s_s[j] = survtime[j];
        float th = theta[j];
        s_e[j] = (_Float16)(1.0f / (1.0f + expf(-th)));   // sigmoid, f32 then round
    }
    __syncthreads();

    const int lane = tid & 31;
    const int wave = tid >> 5;
    const int tile = blockIdx.x * (blockDim.x >> 5) + wave;
    if (tile * 16 >= n) return;                 // wave-uniform; EXEC stays all-1s

    const int M    = lane & 15;                 // A-matrix row owned by this lane
    const int half = lane >> 4;                 // lane group selects K sub-ranges

    const float srow = s_s[tile * 16 + M];

    v16h ones;
#pragma unroll
    for (int e = 0; e < 16; ++e) ones[e] = (_Float16)1.0f;

    v8f acc = {};

    for (int jb = 0; jb < n; jb += 32) {
        // A-layout (documented): element e -> K = 8*half + e      (e = 0..7)
        //                        element e -> K = 16+8*half+(e-8) (e = 8..15)
        const int kA = jb + 8 * half;
        const int kB = jb + 16 + 8 * half;

        // 16B-aligned vector LDS loads (jb%32==0, half-offsets are multiples of 8)
        v4f sA0 = *reinterpret_cast<const v4f*>(s_s + kA);
        v4f sA1 = *reinterpret_cast<const v4f*>(s_s + kA + 4);
        v4f sB0 = *reinterpret_cast<const v4f*>(s_s + kB);
        v4f sB1 = *reinterpret_cast<const v4f*>(s_s + kB + 4);
        v8h eA  = *reinterpret_cast<const v8h*>(s_e + kA);
        v8h eB  = *reinterpret_cast<const v8h*>(s_e + kB);

        v16h a;
#pragma unroll
        for (int e = 0; e < 4; ++e) {
            a[e]      = (srow <= sA0[e]) ? eA[e]     : (_Float16)0.0f;
            a[4 + e]  = (srow <= sA1[e]) ? eA[4 + e] : (_Float16)0.0f;
            a[8 + e]  = (srow <= sB0[e]) ? eB[e]     : (_Float16)0.0f;
            a[12 + e] = (srow <= sB1[e]) ? eB[4 + e] : (_Float16)0.0f;
        }

        // D = A x ones + C  -> row sums of A accumulated in f32
        acc = __builtin_amdgcn_wmma_f32_16x16x32_f16(
            /*neg_a=*/false, a, /*neg_b=*/false, ones,
            /*c_mod=*/(short)0, acc, /*reuse_a=*/false, /*reuse_b=*/false);
    }

    // D layout (documented): lane 0 -> M=0..7 in v0..7, lane 16 -> M=8..15.
    if (M == 0) {
#pragma unroll
        for (int v = 0; v < 8; ++v) {
            const int i   = tile * 16 + 8 * half + v;
            const float e32 = 1.0f / (1.0f + expf(-theta[i]));
            terms[i] = (logf(e32) - logf(acc[v])) * censor[i];
        }
    }
}

// Deterministic fixed-order reduction: strided register sums + LDS tree.
__global__ void __launch_bounds__(256)
cox_reduce_kernel(const float* __restrict__ terms, float* __restrict__ out, int n)
{
    __shared__ float red[256];
    float sum = 0.0f;
    for (int i = threadIdx.x; i < n; i += 256) sum += terms[i];
    red[threadIdx.x] = sum;
    __syncthreads();
#pragma unroll
    for (int s = 128; s > 0; s >>= 1) {
        if ((int)threadIdx.x < s) red[threadIdx.x] += red[threadIdx.x + s];
        __syncthreads();
    }
    if (threadIdx.x == 0) out[0] = -red[0] / (float)n;
}

extern "C" void kernel_launch(void* const* d_in, const int* in_sizes, int n_in,
                              void* d_out, int out_size, void* d_ws, size_t ws_size,
                              hipStream_t stream) {
    (void)n_in; (void)out_size; (void)ws_size;
    const float* survtime = (const float*)d_in[0];
    const float* censor   = (const float*)d_in[1];
    const float* theta    = (const float*)d_in[2];   // hazard_pred, [N,1] flat
    float* out   = (float*)d_out;
    float* terms = (float*)d_ws;                     // N f32 scratch

    const int n = in_sizes[0];                       // 16384
    const int wavesPerBlock = 8;                     // 256 threads = 8 wave32
    const int tiles  = n / 16;
    const int blocks = (tiles + wavesPerBlock - 1) / wavesPerBlock;
    const size_t shmem = (size_t)n * 4u + (size_t)n * 2u;  // f32 s + f16 e = 96KB

    cox_risk_wmma_kernel<<<blocks, 256, shmem, stream>>>(survtime, censor, theta,
                                                         terms, n);
    cox_reduce_kernel<<<1, 256, 0, stream>>>(terms, out, n);
}
